// GraphMambaLayer_68934225101497
// MI455X (gfx1250) — matrix operand order repaired and checked by reading, not compiled, not run on hardware
//
#include <hip/hip_runtime.h>
#include <hip/hip_bf16.h>
#include <math.h>

// ---------------------------------------------------------------------------
// GraphMamba layer for MI455X (gfx1250, wave32, WMMA).
//
// Math note: the reference's argsort/gather/flip/scatter wraps purely
// token-pointwise ops, so gather -> F -> inverse-gather is an exact identity
// and flip -> F -> flip == F.  The whole model collapses to per-token GEMM
// chains over M = B*K = 32768 rows; no sorting is needed (bit-exact).
// ---------------------------------------------------------------------------

#define NLAYERS 2
#define BATCH   16
#define SEQ     2048
#define DMODEL  512
#define DINNER  1024
#define MROWS   (BATCH * SEQ)      // 32768
#define LN_EPS  1e-5f

// GEMM tiling: block 256 threads = 8 waves; block tile 128x128, K-step 32.
// Wave tile 32x64 -> 8 WMMA per k-step per wave (2 A-frags x 4 B-frags).
#define BM 128
#define BN 128
#define BK 32
#define PITCH 40                   // LDS row pitch in bf16 elems (32 + 8 pad)

typedef __attribute__((ext_vector_type(16))) __bf16 bf16x16;
typedef __attribute__((ext_vector_type(8)))  float  floatx8;
typedef __attribute__((ext_vector_type(4)))  int    v4i;
typedef unsigned short ushort_t;

// CDNA5 async global->LDS copies (ASYNCcnt-tracked, no VGPR round trip).
#if defined(__has_builtin)
#if __has_builtin(__builtin_amdgcn_global_load_async_to_lds_b128) && \
    __has_builtin(__builtin_amdgcn_s_wait_asynccnt)
#define USE_ASYNC_LDS 1
#endif
#endif
#ifndef USE_ASYNC_LDS
#define USE_ASYNC_LDS 0
#endif

#if USE_ASYNC_LDS
// Builtin signature (per hipcc diagnostic): param0 = int4 __device__* (global),
// param1 = LDS pointer; pass generic int4* and let clang's builtin pointer
// rewriting insert the address-space casts.
__device__ __forceinline__ void g2l_async16(const __bf16* g, __bf16* l) {
  __builtin_amdgcn_global_load_async_to_lds_b128((v4i*)g, (v4i*)l, 0, 0);
}
#endif

// Epilogue modes
constexpr int EP_BF16       = 0;   // out_bf16 = acc + bias
constexpr int EP_GELU_BF16  = 1;   // out_bf16 = gelu(acc + bias)
constexpr int EP_RES_BF16   = 2;   // out_bf16 = acc + bias + res
constexpr int EP_RES_F32    = 3;   // out_f32  = acc + bias + res
constexpr int EP_RES_KM_F32 = 4;   // out_f32  = (acc + bias + res) * km[row]

__device__ __forceinline__ float bf2f(ushort_t u) {
  union { unsigned int i; float f; } c; c.i = ((unsigned int)u) << 16; return c.f;
}

// ---------------------------------------------------------------------------
// fp32 -> bf16 conversion (weights, once per launch)
// ---------------------------------------------------------------------------
__global__ void f2bf_kernel(const float* __restrict__ x, __bf16* __restrict__ y, int n) {
  int i = blockIdx.x * blockDim.x + threadIdx.x;
  if (i < n) y[i] = (__bf16)x[i];
}

// ---------------------------------------------------------------------------
// WMMA GEMM:  out[M, N] = A[M, K] (bf16) @ W[N, K]^T (bf16) + bias, + epilogue
// grid = (N/BN, M/BM), block = 256.  lda == K.  res is fp32 [M, 512].
// Double-buffered LDS; async tensor-path staging when available.
// ---------------------------------------------------------------------------
template <int MODE>
__global__ __launch_bounds__(256) void gemm_wmma_kernel(
    const __bf16* __restrict__ A, const __bf16* __restrict__ W,
    const float* __restrict__ bias, void* __restrict__ out, int ldc, int col0,
    const float* __restrict__ res, const float* __restrict__ kmv, int K) {
  __shared__ __attribute__((aligned(16))) __bf16 lA[2 * BM * PITCH];
  __shared__ __attribute__((aligned(16))) __bf16 lB[2 * BN * PITCH];

  const int tid  = threadIdx.x;
  const int lane = tid & 31;
  const int wave = tid >> 5;
  const int m_blk = blockIdx.y * BM;
  const int n_blk = blockIdx.x * BN;
  const int wm = (wave >> 1) * 32;     // wave tile rows: 0/32/64/96
  const int wn = (wave & 1) * 64;      // wave tile cols: 0/64
  const int half = lane >> 4;          // 0: lanes 0-15, 1: lanes 16-31
  const int lcol = lane & 15;

  // cooperative-load indices (uint4 = 8 bf16); each tile row pair r, r+64
  const int lr = tid >> 2;
  const int lc = (tid & 3) * 8;
  const __bf16* gA0 = A + (size_t)(m_blk + lr) * K + lc;
  const __bf16* gA1 = A + (size_t)(m_blk + lr + 64) * K + lc;
  const __bf16* gB0 = W + (size_t)(n_blk + lr) * K + lc;
  const __bf16* gB1 = W + (size_t)(n_blk + lr + 64) * K + lc;

  floatx8 acc[2][4];
#pragma unroll
  for (int i = 0; i < 2; ++i)
#pragma unroll
    for (int j = 0; j < 4; ++j) acc[i][j] = (floatx8)0.0f;

  // ---- prologue: stage K-slab 0 into buffer 0 ----------------------------
#if USE_ASYNC_LDS
  g2l_async16(gA0, lA + lr * PITCH + lc);
  g2l_async16(gA1, lA + (lr + 64) * PITCH + lc);
  g2l_async16(gB0, lB + lr * PITCH + lc);
  g2l_async16(gB1, lB + (lr + 64) * PITCH + lc);
#else
  {
    uint4 a0 = *(const uint4*)gA0, a1 = *(const uint4*)gA1;
    uint4 b0 = *(const uint4*)gB0, b1 = *(const uint4*)gB1;
    *(uint4*)(lA + lr * PITCH + lc) = a0;
    *(uint4*)(lA + (lr + 64) * PITCH + lc) = a1;
    *(uint4*)(lB + lr * PITCH + lc) = b0;
    *(uint4*)(lB + (lr + 64) * PITCH + lc) = b1;
  }
#endif

  int cur = 0;
  for (int k0 = 0; k0 < K; k0 += BK, cur ^= 1) {
#if USE_ASYNC_LDS
    __builtin_amdgcn_s_wait_asynccnt(0);
#endif
    __syncthreads();
    const bool has_next = (k0 + BK) < K;
    __bf16* nA = lA + (cur ^ 1) * (BM * PITCH);
    __bf16* nB = lB + (cur ^ 1) * (BN * PITCH);

#if USE_ASYNC_LDS
    if (has_next) {   // async copies of the next K-slab overlap the WMMAs
      g2l_async16(gA0 + k0 + BK, nA + lr * PITCH + lc);
      g2l_async16(gA1 + k0 + BK, nA + (lr + 64) * PITCH + lc);
      g2l_async16(gB0 + k0 + BK, nB + lr * PITCH + lc);
      g2l_async16(gB1 + k0 + BK, nB + (lr + 64) * PITCH + lc);
    }
#else
    uint4 a0, a1, b0, b1;
    if (has_next) {
      a0 = *(const uint4*)(gA0 + k0 + BK);
      a1 = *(const uint4*)(gA1 + k0 + BK);
      b0 = *(const uint4*)(gB0 + k0 + BK);
      b1 = *(const uint4*)(gB1 + k0 + BK);
      if (k0 + 2 * BK < K) {
        __builtin_prefetch(gA0 + k0 + 2 * BK, 0, 2);
        __builtin_prefetch(gB0 + k0 + 2 * BK, 0, 2);
      }
    }
#endif

    const __bf16* bufA = lA + cur * (BM * PITCH);
    const __bf16* bufB = lB + cur * (BN * PITCH);
#pragma unroll
    for (int i = 0; i < 2; ++i) {
      // A frag (16x32 bf16): lane holds row M=lcol, K = half*8 + {0..7,16..23}
      union { bf16x16 v; uint4 q[2]; } af;
      const __bf16* pa = bufA + (wm + i * 16 + lcol) * PITCH + half * 8;
      af.q[0] = *(const uint4*)(pa);
      af.q[1] = *(const uint4*)(pa + 16);
#pragma unroll
      for (int j = 0; j < 4; ++j) {
        // B frag (32x16 bf16): lane holds col N=lcol, K = half*8 + {0..7,16..23}
        union { bf16x16 v; uint4 q[2]; } bfr;
        const __bf16* pb = bufB + (wn + j * 16 + lcol) * PITCH + half * 8;
        bfr.q[0] = *(const uint4*)(pb);
        bfr.q[1] = *(const uint4*)(pb + 16);
        acc[i][j] = __builtin_amdgcn_wmma_f32_16x16x32_bf16(
            false, af.v, false, bfr.v, (short)0, acc[i][j], false, false);
      }
    }

#if !USE_ASYNC_LDS
    if (has_next) {
      *(uint4*)(nA + lr * PITCH + lc) = a0;
      *(uint4*)(nA + (lr + 64) * PITCH + lc) = a1;
      *(uint4*)(nB + lr * PITCH + lc) = b0;
      *(uint4*)(nB + (lr + 64) * PITCH + lc) = b1;
    }
#endif
  }

  // C/D layout: VGPR e, lanes 0-15 -> M=e, N=lane; lanes 16-31 -> M=e+8.
#pragma unroll
  for (int i = 0; i < 2; ++i)
#pragma unroll
    for (int j = 0; j < 4; ++j)
#pragma unroll
      for (int e = 0; e < 8; ++e) {
        const int row  = m_blk + wm + i * 16 + e + 8 * half;
        const int ncol = n_blk + wn + j * 16 + lcol;
        float v = acc[i][j][e] + bias[ncol];
        if constexpr (MODE == EP_GELU_BF16)
          v = 0.5f * v * (1.0f + erff(v * 0.70710678118654752f));
        if constexpr (MODE == EP_RES_BF16 || MODE == EP_RES_F32 || MODE == EP_RES_KM_F32)
          v += res[(size_t)row * DMODEL + ncol];
        if constexpr (MODE == EP_RES_KM_F32) v *= kmv[row];
        if constexpr (MODE == EP_BF16 || MODE == EP_GELU_BF16 || MODE == EP_RES_BF16)
          ((__bf16*)out)[(size_t)row * ldc + col0 + ncol] = (__bf16)v;
        else
          ((float*)out)[(size_t)row * ldc + col0 + ncol] = v;
      }
}

// ---------------------------------------------------------------------------
// Dual LayerNorm: one wave per row of X[M,512]; O0 = bf16(LN(x; w0,b0)),
// O1 = bf16(LN(x; w1,b1)).  Same row stats for both.
// ---------------------------------------------------------------------------
struct bf4 { __bf16 a, b, c, d; };

__global__ __launch_bounds__(256) void ln2_kernel(
    const float* __restrict__ X, const float* __restrict__ w0,
    const float* __restrict__ b0, const float* __restrict__ w1,
    const float* __restrict__ b1, __bf16* __restrict__ O0,
    __bf16* __restrict__ O1) {
  const int lane = threadIdx.x & 31;
  const int row  = blockIdx.x * 8 + (threadIdx.x >> 5);
  const float* x = X + (size_t)row * DMODEL;
  float v[16], s1 = 0.f, s2 = 0.f;
#pragma unroll
  for (int i = 0; i < 4; ++i) {
    float4 t = ((const float4*)x)[lane + 32 * i];
    v[4*i] = t.x; v[4*i+1] = t.y; v[4*i+2] = t.z; v[4*i+3] = t.w;
    s1 += t.x + t.y + t.z + t.w;
    s2 += t.x*t.x + t.y*t.y + t.z*t.z + t.w*t.w;
  }
#pragma unroll
  for (int o = 16; o > 0; o >>= 1) { s1 += __shfl_xor(s1, o, 32); s2 += __shfl_xor(s2, o, 32); }
  const float mean = s1 * (1.0f / DMODEL);
  const float rstd = rsqrtf(s2 * (1.0f / DMODEL) - mean * mean + LN_EPS);
#pragma unroll
  for (int i = 0; i < 4; ++i) {
    const int idx = 4 * (lane + 32 * i);
    bf4 o0, o1;
    float n0 = (v[4*i]   - mean) * rstd, n1 = (v[4*i+1] - mean) * rstd;
    float n2 = (v[4*i+2] - mean) * rstd, n3 = (v[4*i+3] - mean) * rstd;
    o0.a = (__bf16)(n0 * w0[idx]   + b0[idx]);
    o0.b = (__bf16)(n1 * w0[idx+1] + b0[idx+1]);
    o0.c = (__bf16)(n2 * w0[idx+2] + b0[idx+2]);
    o0.d = (__bf16)(n3 * w0[idx+3] + b0[idx+3]);
    o1.a = (__bf16)(n0 * w1[idx]   + b1[idx]);
    o1.b = (__bf16)(n1 * w1[idx+1] + b1[idx+1]);
    o1.c = (__bf16)(n2 * w1[idx+2] + b1[idx+2]);
    o1.d = (__bf16)(n3 * w1[idx+3] + b1[idx+3]);
    *(bf4*)(O0 + (size_t)row * DMODEL + idx) = o0;
    *(bf4*)(O1 + (size_t)row * DMODEL + idx) = o1;
  }
}

// ---------------------------------------------------------------------------
// Gate + inner LN: ZG[M,2048] bf16 (z | g halves) -> U[M,1024] bf16
// u = LN( silu(z) * sigmoid(g) ; w,b ).  One wave per row (32 elems/lane).
// ---------------------------------------------------------------------------
__global__ __launch_bounds__(256) void act_iln_kernel(
    const __bf16* __restrict__ ZG, const float* __restrict__ w,
    const float* __restrict__ b, __bf16* __restrict__ U) {
  const int lane = threadIdx.x & 31;
  const int row  = blockIdx.x * 8 + (threadIdx.x >> 5);
  const ushort_t* zrow = (const ushort_t*)ZG + (size_t)row * (2 * DINNER);
  float u[32], s1 = 0.f, s2 = 0.f;
#pragma unroll
  for (int i = 0; i < 4; ++i) {
    ushort_t zz[8], gg[8];
    *(uint4*)zz = ((const uint4*)zrow)[lane + 32 * i];
    *(uint4*)gg = ((const uint4*)(zrow + DINNER))[lane + 32 * i];
#pragma unroll
    for (int c = 0; c < 8; ++c) {
      const float z = bf2f(zz[c]), g = bf2f(gg[c]);
      const float val = (z / (1.0f + __expf(-z))) * (1.0f / (1.0f + __expf(-g)));
      u[8*i + c] = val; s1 += val; s2 += val * val;
    }
  }
#pragma unroll
  for (int o = 16; o > 0; o >>= 1) { s1 += __shfl_xor(s1, o, 32); s2 += __shfl_xor(s2, o, 32); }
  const float mean = s1 * (1.0f / DINNER);
  const float rstd = rsqrtf(s2 * (1.0f / DINNER) - mean * mean + LN_EPS);
#pragma unroll
  for (int i = 0; i < 4; ++i) {
    union { __bf16 h[8]; uint4 q; } o8;
    const int idx = 8 * (lane + 32 * i);
#pragma unroll
    for (int c = 0; c < 8; ++c)
      o8.h[c] = (__bf16)((u[8*i + c] - mean) * rstd * w[idx + c] + b[idx + c]);
    ((uint4*)(U + (size_t)row * DINNER))[lane + 32 * i] = o8.q;
  }
}

// ---------------------------------------------------------------------------
// Chained LN: S = LN(X; nw,nb) (fp32 out), H = bf16(LN(S; fw,fb)).
// ---------------------------------------------------------------------------
__global__ __launch_bounds__(256) void ln_chain_kernel(
    const float* __restrict__ X, const float* __restrict__ nw,
    const float* __restrict__ nb, const float* __restrict__ fw,
    const float* __restrict__ fb, float* __restrict__ S,
    __bf16* __restrict__ H) {
  const int lane = threadIdx.x & 31;
  const int row  = blockIdx.x * 8 + (threadIdx.x >> 5);
  const float* x = X + (size_t)row * DMODEL;
  float v[16], s1 = 0.f, s2 = 0.f;
#pragma unroll
  for (int i = 0; i < 4; ++i) {
    float4 t = ((const float4*)x)[lane + 32 * i];
    v[4*i] = t.x; v[4*i+1] = t.y; v[4*i+2] = t.z; v[4*i+3] = t.w;
    s1 += t.x + t.y + t.z + t.w;
    s2 += t.x*t.x + t.y*t.y + t.z*t.z + t.w*t.w;
  }
#pragma unroll
  for (int o = 16; o > 0; o >>= 1) { s1 += __shfl_xor(s1, o, 32); s2 += __shfl_xor(s2, o, 32); }
  float mean = s1 * (1.0f / DMODEL);
  float rstd = rsqrtf(s2 * (1.0f / DMODEL) - mean * mean + LN_EPS);
  float t1 = 0.f, t2 = 0.f;
#pragma unroll
  for (int i = 0; i < 4; ++i)
#pragma unroll
    for (int c = 0; c < 4; ++c) {
      const int idx = 4 * (lane + 32 * i) + c;
      const float tv = (v[4*i + c] - mean) * rstd * nw[idx] + nb[idx];
      v[4*i + c] = tv; t1 += tv; t2 += tv * tv;
    }
#pragma unroll
  for (int i = 0; i < 4; ++i)
    ((float4*)(S + (size_t)row * DMODEL))[lane + 32 * i] =
        make_float4(v[4*i], v[4*i+1], v[4*i+2], v[4*i+3]);
#pragma unroll
  for (int o = 16; o > 0; o >>= 1) { t1 += __shfl_xor(t1, o, 32); t2 += __shfl_xor(t2, o, 32); }
  const float m2 = t1 * (1.0f / DMODEL);
  const float r2 = rsqrtf(t2 * (1.0f / DMODEL) - m2 * m2 + LN_EPS);
#pragma unroll
  for (int i = 0; i < 4; ++i) {
    const int idx = 4 * (lane + 32 * i);
    bf4 oh;
    oh.a = (__bf16)((v[4*i]   - m2) * r2 * fw[idx]   + fb[idx]);
    oh.b = (__bf16)((v[4*i+1] - m2) * r2 * fw[idx+1] + fb[idx+1]);
    oh.c = (__bf16)((v[4*i+2] - m2) * r2 * fw[idx+2] + fb[idx+2]);
    oh.d = (__bf16)((v[4*i+3] - m2) * r2 * fw[idx+3] + fb[idx+3]);
    *(bf4*)(H + (size_t)row * DMODEL + idx) = oh;
  }
}

// ---------------------------------------------------------------------------
// Host launcher
// ---------------------------------------------------------------------------
extern "C" void kernel_launch(void* const* d_in, const int* in_sizes, int n_in,
                              void* d_out, int out_size, void* d_ws, size_t ws_size,
                              hipStream_t stream) {
  (void)in_sizes; (void)n_in; (void)out_size; (void)ws_size;
  const float* s_in     = (const float*)d_in[0];
  const float* km       = (const float*)d_in[1];
  const float* f_ln_w   = (const float*)d_in[2];
  const float* f_ln_b   = (const float*)d_in[3];
  const float* f_pin_w  = (const float*)d_in[4];
  const float* f_pin_b  = (const float*)d_in[5];
  const float* f_pout_w = (const float*)d_in[6];
  const float* f_pout_b = (const float*)d_in[7];
  const float* f_iln_w  = (const float*)d_in[8];
  const float* f_iln_b  = (const float*)d_in[9];
  const float* b_ln_w   = (const float*)d_in[10];
  const float* b_ln_b   = (const float*)d_in[11];
  const float* b_pin_w  = (const float*)d_in[12];
  const float* b_pin_b  = (const float*)d_in[13];
  const float* b_pout_w = (const float*)d_in[14];
  const float* b_pout_b = (const float*)d_in[15];
  const float* b_iln_w  = (const float*)d_in[16];
  const float* b_iln_b  = (const float*)d_in[17];
  const float* mrg_w    = (const float*)d_in[18];
  const float* mrg_b    = (const float*)d_in[19];
  const float* nrm_w    = (const float*)d_in[20];
  const float* nrm_b    = (const float*)d_in[21];
  const float* ffn_ln_w = (const float*)d_in[22];
  const float* ffn_ln_b = (const float*)d_in[23];
  const float* ffn_w1   = (const float*)d_in[24];
  const float* ffn_b1   = (const float*)d_in[25];
  const float* ffn_w2   = (const float*)d_in[26];
  const float* ffn_b2   = (const float*)d_in[27];
  float* s = (float*)d_out;     // state lives in d_out, updated in place

  // ---- workspace carve-up (~360 MB) --------------------------------------
  char* base = (char*)d_ws;
  size_t off = 0;
  auto take = [&](size_t bytes) -> char* {
    char* p = base + off; off += (bytes + 255) & ~(size_t)255; return p;
  };
  __bf16* WpinF  = (__bf16*)take((size_t)NLAYERS * 2048 * 512 * 2);
  __bf16* WpoutF = (__bf16*)take((size_t)NLAYERS * 512 * 1024 * 2);
  __bf16* WpinB  = (__bf16*)take((size_t)NLAYERS * 2048 * 512 * 2);
  __bf16* WpoutB = (__bf16*)take((size_t)NLAYERS * 512 * 1024 * 2);
  __bf16* Wmrg   = (__bf16*)take((size_t)NLAYERS * 512 * 1024 * 2);
  __bf16* Wf1    = (__bf16*)take((size_t)NLAYERS * 2048 * 512 * 2);
  __bf16* Wf2    = (__bf16*)take((size_t)NLAYERS * 512 * 2048 * 2);
  __bf16* A0 = (__bf16*)take((size_t)MROWS * 512 * 2);
  __bf16* A1 = (__bf16*)take((size_t)MROWS * 512 * 2);
  __bf16* ZG = (__bf16*)take((size_t)MROWS * 2048 * 2);   // also FFN hidden
  __bf16* U  = (__bf16*)take((size_t)MROWS * 1024 * 2);   // aliased as T below
  __bf16* FB = (__bf16*)take((size_t)MROWS * 1024 * 2);
  float*  T  = (float*)U;   // [M,512] fp32; lifetime disjoint from U (bf16)

  auto conv = [&](const float* src, __bf16* dst, int n) {
    f2bf_kernel<<<(n + 255) / 256, 256, 0, stream>>>(src, dst, n);
  };
  conv(f_pin_w,  WpinF,  NLAYERS * 2048 * 512);
  conv(f_pout_w, WpoutF, NLAYERS * 512 * 1024);
  conv(b_pin_w,  WpinB,  NLAYERS * 2048 * 512);
  conv(b_pout_w, WpoutB, NLAYERS * 512 * 1024);
  conv(mrg_w,    Wmrg,   NLAYERS * 512 * 1024);
  conv(ffn_w1,   Wf1,    NLAYERS * 2048 * 512);
  conv(ffn_w2,   Wf2,    NLAYERS * 512 * 2048);

  (void)hipMemcpyAsync(s, s_in, (size_t)MROWS * DMODEL * sizeof(float),
                       hipMemcpyDeviceToDevice, stream);

  const dim3 blk(256);
  const int  LNG = MROWS / 8;                 // wave-per-row grids
  const dim3 gN2048(2048 / BN, MROWS / BM);   // (16, 256)
  const dim3 gN512 (512  / BN, MROWS / BM);   // (4, 256)

  for (int i = 0; i < NLAYERS; ++i) {
    const size_t wPin = (size_t)i * 2048 * 512, wPout = (size_t)i * 512 * 1024;
    const size_t wM = (size_t)i * 512 * 1024, wF1 = (size_t)i * 2048 * 512;
    const size_t wF2 = (size_t)i * 512 * 2048;

    // 1) A0 = bf16(LN(s; fwd_ln)), A1 = bf16(LN(s; bwd_ln))
    ln2_kernel<<<LNG, blk, 0, stream>>>(s, f_ln_w + i * 512, f_ln_b + i * 512,
                                        b_ln_w + i * 512, b_ln_b + i * 512, A0, A1);
    // 2) ZG = A0 @ WpinF^T + b                      [M,2048]
    gemm_wmma_kernel<EP_BF16><<<gN2048, blk, 0, stream>>>(
        A0, WpinF + wPin, f_pin_b + i * 2048, ZG, 2048, 0, nullptr, nullptr, 512);
    // 3) U = LN(silu(z)*sigmoid(g); fwd_iln)        [M,1024]
    act_iln_kernel<<<LNG, blk, 0, stream>>>(ZG, f_iln_w + i * 1024, f_iln_b + i * 1024, U);
    // 4) FB[:,0:512] = bf16(s + U @ WpoutF^T + b)
    gemm_wmma_kernel<EP_RES_BF16><<<gN512, blk, 0, stream>>>(
        U, WpoutF + wPout, f_pout_b + i * 512, FB, 1024, 0, s, nullptr, 1024);
    // 5-7) backward branch (identical math, bwd weights) -> FB[:,512:1024]
    gemm_wmma_kernel<EP_BF16><<<gN2048, blk, 0, stream>>>(
        A1, WpinB + wPin, b_pin_b + i * 2048, ZG, 2048, 0, nullptr, nullptr, 512);
    act_iln_kernel<<<LNG, blk, 0, stream>>>(ZG, b_iln_w + i * 1024, b_iln_b + i * 1024, U);
    gemm_wmma_kernel<EP_RES_BF16><<<gN512, blk, 0, stream>>>(
        U, WpoutB + wPout, b_pout_b + i * 512, FB, 1024, 512, s, nullptr, 1024);
    // 8) T = s + FB @ Wmrg^T + mrg_b                [M,512] fp32 (pre-norm)
    gemm_wmma_kernel<EP_RES_F32><<<gN512, blk, 0, stream>>>(
        FB, Wmrg + wM, mrg_b + i * 512, T, 512, 0, s, nullptr, 1024);
    // 9) s = LN(T; nrm); A0 = bf16(LN(s; ffn_ln))
    ln_chain_kernel<<<LNG, blk, 0, stream>>>(T, nrm_w + i * 512, nrm_b + i * 512,
                                             ffn_ln_w + i * 512, ffn_ln_b + i * 512, s, A0);
    // 10) ZG = gelu(A0 @ Wf1^T + b1)                [M,2048]
    gemm_wmma_kernel<EP_GELU_BF16><<<gN2048, blk, 0, stream>>>(
        A0, Wf1 + wF1, ffn_b1 + i * 2048, ZG, 2048, 0, nullptr, nullptr, 512);
    // 11) s = (s + ZG @ Wf2^T + b2) * km
    gemm_wmma_kernel<EP_RES_KM_F32><<<gN512, blk, 0, stream>>>(
        ZG, Wf2 + wF2, ffn_b2 + i * 512, s, 512, 0, s, km, 2048);
  }
}
